// HyperLayer_43052752175188
// MI455X (gfx1250) — matrix operand order repaired and checked by reading, not compiled, not run on hardware
//
#include <hip/hip_runtime.h>
#include <hip/hip_bf16.h>
#include <stddef.h>

// ---------------------------------------------------------------------------
// HyperLayer on MI455X (gfx1250, wave32, WMMA + Tensor Data Mover).
//
// out[b,o] = sum_{h,i} hg[b,h]*x[b,i]*W2g[h,o*256+i]      (68.7 GFLOP GEMM)
//          + sum_i x[b,i]*b2g[o*256+i] + (hb@W2b)[b,o] + b2b[o]
//
// Khatri-Rao factorization: never materialize the 1 GB per-sample weight
// tensor. bf16 inputs + f32 WMMA accumulation; W2g B-tiles streamed to LDS
// by TDM (double-buffered, overlapped with WMMA); x block TDM-staged once.
// ---------------------------------------------------------------------------

#define IN_F   256
#define OUT_F  256
#define HYPER  128
#define BATCH  4096

typedef __attribute__((ext_vector_type(16))) __bf16        v16bf;
typedef __attribute__((ext_vector_type(8)))  float         v8f;
typedef __attribute__((ext_vector_type(4)))  unsigned int  u32x4;
typedef __attribute__((ext_vector_type(8)))  int           i32x8;
typedef __attribute__((ext_vector_type(4)))  int           i32x4;

union FragU { u32x4 q[2]; v16bf v; };

__device__ __forceinline__ float bf2f(unsigned int lo16) {
    return __builtin_bit_cast(float, lo16 << 16);
}
__device__ __forceinline__ unsigned short f2bf(float f) {
    unsigned int u = __builtin_bit_cast(unsigned int, f);
    u += 0x7FFFu + ((u >> 16) & 1u);           // round-to-nearest-even
    return (unsigned short)(u >> 16);
}

__device__ __forceinline__ v16bf splat_bf(unsigned int hbits) {
    unsigned int p = (hbits & 0xFFFFu) | (hbits << 16);
    FragU f;
    u32x4 v; v[0] = p; v[1] = p; v[2] = p; v[3] = p;
    f.q[0] = v; f.q[1] = v;
    return f.v;
}

__device__ __forceinline__ v8f wmma_bf16(const FragU& a, const FragU& b, v8f c) {
    // 8 args: (neg_a, A, neg_b, B, c_mod, C, reuse_a, reuse_b)
    return __builtin_amdgcn_wmma_f32_16x16x32_bf16(false, a.v, false, b.v,
                                                   (short)0, c, false, false);
}

// --------------------------- TDM 2D tile load ------------------------------
// D# per CDNA5 ISA: count=1, type=2 ("image"),
// data_size=2B, pad_enable, tile==tensor dims (no OOB), 2D only.
// 6-arg builtin (clang-23/therock-10.0): (v4u g0, v8i g1, v4i g2, v4i g3,
// v8i g4, i32 cpol); groups 2/3/4 unused for 2D tiles -> zeros.
__device__ __forceinline__ void tdm_load_2d(unsigned lds_off, const void* gptr,
                                            unsigned tile_d0, unsigned tile_d1,
                                            unsigned stride_d0,
                                            unsigned pad_int_code,
                                            unsigned pad_amt_code) {
    unsigned long long ga = (unsigned long long)gptr;
    u32x4 g0;
    g0[0] = 1u;                                        // count=1, user mode
    g0[1] = lds_off;                                   // lds_addr [63:32]
    g0[2] = (unsigned)(ga & 0xFFFFFFFFu);              // global_addr [95:64]
    g0[3] = (unsigned)((ga >> 32) & 0x01FFFFFFu)       // global_addr [120:96]
          | (2u << 30);                                // type=2
    i32x8 g1;
    g1[0] = (int)((1u << 16)                           // data_size=2B
                | (1u << 20)                           // pad_enable
                | (pad_int_code << 22)                 // pad_interval
                | (pad_amt_code << 25));               // pad_amount
    g1[1] = (int)((tile_d0 & 0xFFFFu) << 16);          // tensor_dim0[15:0]
    g1[2] = (int)((tile_d1 & 0xFFFFu) << 16);          // tensor_dim1[15:0]
    g1[3] = (int)((tile_d0 & 0xFFFFu) << 16);          // tile_dim0
    g1[4] = (int)(tile_d1 & 0xFFFFu);                  // tile_dim1 (dim2=0)
    g1[5] = (int)stride_d0;                            // tensor_dim0_stride
    g1[6] = 0;
    g1[7] = 0;
    i32x4 z4 = {0, 0, 0, 0};
    i32x8 z8 = {0, 0, 0, 0, 0, 0, 0, 0};
    __builtin_amdgcn_tensor_load_to_lds(g0, g1, z4, z4, z8, 0);
}

// ------------------------- tile geometry -----------------------------------
#define BM   128
#define BN   64
#define BK   32
#define LDA  40   // 32 data + 8 pad ushorts -> 80B rows (16B aligned,
#define LDB  40   //   conflict-free b128 reads; == TDM pad 16DW/4DW)
#define XLD  264  // 256 data + 8 pad -> 528B rows (== TDM pad 128DW/4DW)

// Fragment layouts per CDNA5 ISA 7.12.2 (wave32):
//  A 16x32 bf16: lane l = row M=l&15; K chunks {c*8.., 16+c*8..}, c=l>>4
//  B 32x16 bf16: lane l = col N=l&15; K chunks {c*8.., 16+c*8..}, c=l>>4
__device__ __forceinline__ void load_afrag(FragU& f, const unsigned short (*As)[LDA],
                                           int rowBase, int lane) {
    int r  = rowBase + (lane & 15);
    int kc = (lane >> 4) * 8;
    f.q[0] = *(const u32x4*)&As[r][kc];
    f.q[1] = *(const u32x4*)&As[r][16 + kc];
}
__device__ __forceinline__ void load_bfrag(FragU& f, const unsigned short (*Bs)[LDB],
                                           int colBase, int lane) {
    int c  = colBase + (lane & 15);
    int kh = (lane >> 4) * 8;
    f.q[0] = *(const u32x4*)&Bs[c][kh];
    f.q[1] = *(const u32x4*)&Bs[c][kh + 16];
}

// ------------------------- prep kernels ------------------------------------
__global__ void k_cvt_bf16(unsigned short* __restrict__ dst,
                           const float* __restrict__ src, int n) {
    for (int i = blockIdx.x * blockDim.x + threadIdx.x; i < n;
         i += gridDim.x * blockDim.x)
        dst[i] = f2bf(src[i]);
}

__global__ void k_build_wcatT(unsigned short* __restrict__ dst,
                              const float* __restrict__ W1g,
                              const float* __restrict__ W1b) {
    int idx = blockIdx.x * blockDim.x + threadIdx.x;   // 65536 threads
    int n = idx >> 8, k = idx & 255;
    float v = (n < HYPER) ? W1g[k * HYPER + n] : W1b[k * HYPER + (n - HYPER)];
    dst[idx] = f2bf(v);
}

__global__ void k_build_bcat(float* __restrict__ dst,
                             const float* __restrict__ b1g,
                             const float* __restrict__ b1b) {
    int n = threadIdx.x;                               // 256 threads
    dst[n] = (n < HYPER) ? b1g[n] : b1b[n - HYPER];
}

__global__ void k_build_bbiasT(unsigned short* __restrict__ dst,
                               const float* __restrict__ W2b,
                               const float* __restrict__ b2g) {
    int idx = blockIdx.x * blockDim.x + threadIdx.x;   // 98304 threads
    int n = idx / 384, k = idx % 384;
    float v = (k < HYPER) ? W2b[k * OUT_F + n] : b2g[n * IN_F + (k - HYPER)];
    dst[idx] = f2bf(v);
}

// ------------------------- hidden GEMM: Hcat = relu(x @ Wcat + bcat) --------
__global__ __launch_bounds__(256)
void k_hidden(const unsigned short* __restrict__ xbf,
              const unsigned short* __restrict__ wcatT,
              const float* __restrict__ bcat,
              unsigned short* __restrict__ hg,
              unsigned short* __restrict__ hb) {
    __shared__ __align__(16) unsigned short As[BM][LDA];
    __shared__ __align__(16) unsigned short Bs[BN][LDB];
    const int t = threadIdx.x;
    const int mBase = blockIdx.y * BM, nBase = blockIdx.x * BN;
    const int lane = t & 31, wid = t >> 5;
    const int wm = (wid & 3) * 32, wn = (wid >> 2) * 32;
    const int am = t & 127, ah = (t >> 7) * 16;
    const int bn = t & 63,  bp = (t >> 6) * 8;

    v8f zero = {0.f,0.f,0.f,0.f,0.f,0.f,0.f,0.f};
    v8f acc[2][2] = {{zero, zero}, {zero, zero}};

    for (int kt = 0; kt < IN_F / BK; ++kt) {
        const int kBase = kt * BK;
        const unsigned short* xs = xbf + (size_t)(mBase + am) * IN_F + kBase + ah;
        *(u32x4*)&As[am][ah]     = *(const u32x4*)xs;
        *(u32x4*)&As[am][ah + 8] = *(const u32x4*)(xs + 8);
        const unsigned short* ws = wcatT + (size_t)(nBase + bn) * IN_F + kBase + bp;
        *(u32x4*)&Bs[bn][bp] = *(const u32x4*)ws;
        __syncthreads();

        FragU af[2], bfv[2];
        #pragma unroll
        for (int mi = 0; mi < 2; ++mi) load_afrag(af[mi], As, wm + mi * 16, lane);
        #pragma unroll
        for (int ni = 0; ni < 2; ++ni) load_bfrag(bfv[ni], Bs, wn + ni * 16, lane);
        #pragma unroll
        for (int mi = 0; mi < 2; ++mi)
            #pragma unroll
            for (int ni = 0; ni < 2; ++ni)
                acc[mi][ni] = wmma_bf16(af[mi], bfv[ni], acc[mi][ni]);
        __syncthreads();
    }

    const int cl = lane & 15, rh = (lane >> 4) * 8;
    #pragma unroll
    for (int mi = 0; mi < 2; ++mi)
        #pragma unroll
        for (int ni = 0; ni < 2; ++ni) {
            int col = nBase + wn + ni * 16 + cl;
            float bc = bcat[col];
            #pragma unroll
            for (int v = 0; v < 8; ++v) {
                int row = mBase + wm + mi * 16 + rh + v;
                float h = acc[mi][ni][v] + bc;
                h = h > 0.f ? h : 0.f;
                unsigned short hv = f2bf(h);
                if (col < HYPER) hg[(size_t)row * HYPER + col] = hv;
                else             hb[(size_t)row * HYPER + (col - HYPER)] = hv;
            }
        }
}

// ---------------- bias GEMM: biasacc = [hb|x] @ BbiasT' + b2b ---------------
__global__ __launch_bounds__(256)
void k_biaspart(const unsigned short* __restrict__ hb,
                const unsigned short* __restrict__ xbf,
                const unsigned short* __restrict__ bbiasT,
                const float* __restrict__ b2b,
                float* __restrict__ biasacc) {
    __shared__ __align__(16) unsigned short As[BM][LDA];
    __shared__ __align__(16) unsigned short Bs[BN][LDB];
    const int t = threadIdx.x;
    const int mBase = blockIdx.y * BM, nBase = blockIdx.x * BN;
    const int lane = t & 31, wid = t >> 5;
    const int wm = (wid & 3) * 32, wn = (wid >> 2) * 32;
    const int am = t & 127, ah = (t >> 7) * 16;
    const int bn = t & 63,  bp = (t >> 6) * 8;

    v8f zero = {0.f,0.f,0.f,0.f,0.f,0.f,0.f,0.f};
    v8f acc[2][2] = {{zero, zero}, {zero, zero}};

    for (int kt = 0; kt < 384 / BK; ++kt) {
        const int kBase = kt * BK;
        const unsigned short* asrc =
            (kBase < HYPER)
                ? hb  + (size_t)(mBase + am) * HYPER + kBase + ah
                : xbf + (size_t)(mBase + am) * IN_F  + (kBase - HYPER) + ah;
        *(u32x4*)&As[am][ah]     = *(const u32x4*)asrc;
        *(u32x4*)&As[am][ah + 8] = *(const u32x4*)(asrc + 8);
        const unsigned short* bsrc = bbiasT + (size_t)(nBase + bn) * 384 + kBase + bp;
        *(u32x4*)&Bs[bn][bp] = *(const u32x4*)bsrc;
        __syncthreads();

        FragU af[2], bfv[2];
        #pragma unroll
        for (int mi = 0; mi < 2; ++mi) load_afrag(af[mi], As, wm + mi * 16, lane);
        #pragma unroll
        for (int ni = 0; ni < 2; ++ni) load_bfrag(bfv[ni], Bs, wn + ni * 16, lane);
        #pragma unroll
        for (int mi = 0; mi < 2; ++mi)
            #pragma unroll
            for (int ni = 0; ni < 2; ++ni)
                acc[mi][ni] = wmma_bf16(af[mi], bfv[ni], acc[mi][ni]);
        __syncthreads();
    }

    const int cl = lane & 15, rh = (lane >> 4) * 8;
    #pragma unroll
    for (int mi = 0; mi < 2; ++mi)
        #pragma unroll
        for (int ni = 0; ni < 2; ++ni) {
            int col = nBase + wn + ni * 16 + cl;
            float bb = b2b[col];
            #pragma unroll
            for (int v = 0; v < 8; ++v) {
                int row = mBase + wm + mi * 16 + rh + v;
                biasacc[(size_t)row * OUT_F + col] = acc[mi][ni][v] + bb;
            }
        }
}

// ---------------- main Khatri-Rao GEMM (TDM double-buffered) ----------------
// out[b,o] = sum_{h,i} (hg[b,h]*x[b,i]) * W2g[h, o*256+i] + biasacc[b,o]
// M=4096 N=256 K=32768. x block TDM-staged in LDS once; A-frag = x-frag
// scaled by the per-lane scalar hg[m,h] (v_pk_mul_bf16); B tiles streamed
// by TDM into a double buffer, overlapped with WMMA.
__global__ __launch_bounds__(256)
void k_main(const unsigned short* __restrict__ xbf,
            const unsigned short* __restrict__ hg,
            const unsigned short* __restrict__ w2gbf,
            const float* __restrict__ biasacc,
            float* __restrict__ out) {
    __shared__ __align__(16) unsigned short Xs[BM][XLD];      // 67,584 B
    __shared__ __align__(16) unsigned short Bs[2][BN][LDB];   // 10,240 B
    const int t = threadIdx.x;
    const int mBase = blockIdx.y * BM, nBase = blockIdx.x * BN;
    const int lane = t & 31, wid = t >> 5;
    const int wm = (wid & 3) * 32, wn = (wid >> 2) * 32;
    const int rsel = lane & 15, kc = (lane >> 4) * 8, kh = (lane >> 4) * 8;

    const unsigned xs_lds = (unsigned)(unsigned long long)(void*)&Xs[0][0];
    const unsigned bs_lds0 = (unsigned)(unsigned long long)(void*)&Bs[0][0][0];
    const unsigned bs_lds1 = (unsigned)(unsigned long long)(void*)&Bs[1][0][0];

    // Prologue: TDM-stage the x block (128x256, pad 4DW per 128DW -> 528B rows)
    // and the first B tile (64x32, pad 4DW per 16DW -> 80B rows).
    if (wid == 0) {
        tdm_load_2d(xs_lds, xbf + (size_t)mBase * IN_F,
                    /*d0=*/IN_F, /*d1=*/BM, /*stride=*/IN_F,
                    /*pad_int=*/6, /*pad_amt=*/3);
        tdm_load_2d(bs_lds0, w2gbf + (size_t)nBase * IN_F,
                    /*d0=*/BK, /*d1=*/BN, /*stride=*/IN_F,
                    /*pad_int=*/3, /*pad_amt=*/3);
    }
    __builtin_amdgcn_s_wait_tensorcnt(0);
    __syncthreads();

    v8f zero = {0.f,0.f,0.f,0.f,0.f,0.f,0.f,0.f};
    v8f acc[2][2] = {{zero, zero}, {zero, zero}};

    int it = 0;                                   // linear K-tile index, 0..1023
    for (int h = 0; h < HYPER; ++h) {
        // per-lane scalar hg[m,h] for this wave's two row blocks (8-iter reuse)
        v16bf sA0 = splat_bf(hg[(size_t)(mBase + wm + rsel) * HYPER + h]);
        v16bf sA1 = splat_bf(hg[(size_t)(mBase + wm + 16 + rsel) * HYPER + h]);
        for (int i0 = 0; i0 < IN_F; i0 += BK) {
            const int nit = it + 1;
            if (nit < HYPER * (IN_F / BK) && wid == 0) {
                const int hn = nit >> 3, i0n = (nit & 7) << 5;
                tdm_load_2d((nit & 1) ? bs_lds1 : bs_lds0,
                            w2gbf + (size_t)hn * (OUT_F * IN_F)
                                  + (size_t)nBase * IN_F + i0n,
                            BK, BN, IN_F, 3, 3);
            }

            FragU af[2], bfv[2];
            // A-frags straight from the staged x block, scaled in registers
            {
                int r0 = wm + rsel, r1 = wm + 16 + rsel;
                af[0].q[0] = *(const u32x4*)&Xs[r0][i0 + kc];
                af[0].q[1] = *(const u32x4*)&Xs[r0][i0 + 16 + kc];
                af[1].q[0] = *(const u32x4*)&Xs[r1][i0 + kc];
                af[1].q[1] = *(const u32x4*)&Xs[r1][i0 + 16 + kc];
                af[0].v = af[0].v * sA0;          // v_pk_mul_bf16 x8
                af[1].v = af[1].v * sA1;
            }
            const unsigned short (*Bcur)[LDB] = Bs[it & 1];
            #pragma unroll
            for (int ni = 0; ni < 2; ++ni) {
                int c = wn + ni * 16 + rsel;
                bfv[ni].q[0] = *(const u32x4*)&Bcur[c][kh];
                bfv[ni].q[1] = *(const u32x4*)&Bcur[c][kh + 16];
            }
            #pragma unroll
            for (int mi = 0; mi < 2; ++mi)
                #pragma unroll
                for (int ni = 0; ni < 2; ++ni)
                    acc[mi][ni] = wmma_bf16(af[mi], bfv[ni], acc[mi][ni]);

            __builtin_amdgcn_s_wait_tensorcnt(0);  // next B tile landed
            __syncthreads();                       // safe to reuse buffers
            ++it;
        }
    }

    const int cl = lane & 15, rh = (lane >> 4) * 8;
    #pragma unroll
    for (int mi = 0; mi < 2; ++mi)
        #pragma unroll
        for (int ni = 0; ni < 2; ++ni) {
            int col = nBase + wn + ni * 16 + cl;
            #pragma unroll
            for (int v = 0; v < 8; ++v) {
                int row = mBase + wm + mi * 16 + rh + v;
                size_t idx = (size_t)row * OUT_F + col;
                out[idx] = acc[mi][ni][v] + biasacc[idx];
            }
        }
}

// ------------------------- workspace layout (bytes) ------------------------
#define XBF_OFF    0u          // 4096*256 bf16   = 2 MiB
#define HG_OFF     2097152u    // 4096*128 bf16   = 1 MiB
#define HB_OFF     3145728u    // 4096*128 bf16   = 1 MiB
#define W2G_OFF    4194304u    // 128*65536 bf16  = 16 MiB
#define WCAT_OFF   20971520u   // 256*256 bf16
#define BCAT_OFF   21102592u   // 256 f32
#define BBIAS_OFF  21103616u   // 256*384 bf16
#define BACC_OFF   21300224u   // 4096*256 f32    = 4 MiB
// total ~24.3 MiB of d_ws

extern "C" void kernel_launch(void* const* d_in, const int* in_sizes, int n_in,
                              void* d_out, int out_size, void* d_ws, size_t ws_size,
                              hipStream_t stream) {
    const float* x   = (const float*)d_in[0];
    const float* W1g = (const float*)d_in[1];
    const float* b1g = (const float*)d_in[2];
    const float* W2g = (const float*)d_in[3];
    const float* b2g = (const float*)d_in[4];
    const float* W1b = (const float*)d_in[5];
    const float* b1b = (const float*)d_in[6];
    const float* W2b = (const float*)d_in[7];
    const float* b2b = (const float*)d_in[8];
    float* out = (float*)d_out;

    char* ws = (char*)d_ws;
    unsigned short* xbf    = (unsigned short*)(ws + XBF_OFF);
    unsigned short* hg     = (unsigned short*)(ws + HG_OFF);
    unsigned short* hb     = (unsigned short*)(ws + HB_OFF);
    unsigned short* w2gbf  = (unsigned short*)(ws + W2G_OFF);
    unsigned short* wcatT  = (unsigned short*)(ws + WCAT_OFF);
    float*          bcat   = (float*)        (ws + BCAT_OFF);
    unsigned short* bbiasT = (unsigned short*)(ws + BBIAS_OFF);
    float*          bacc   = (float*)        (ws + BACC_OFF);

    k_cvt_bf16<<<2048, 256, 0, stream>>>(xbf, x, BATCH * IN_F);
    k_cvt_bf16<<<8192, 256, 0, stream>>>(w2gbf, W2g, HYPER * IN_F * OUT_F);
    k_build_wcatT<<<256, 256, 0, stream>>>(wcatT, W1g, W1b);
    k_build_bcat<<<1, 256, 0, stream>>>(bcat, b1g, b1b);
    k_build_bbiasT<<<384, 256, 0, stream>>>(bbiasT, W2b, b2g);

    dim3 block(256);
    dim3 grid(OUT_F / BN, BATCH / BM);   // (4, 32) -> 128 workgroups
    k_hidden  <<<grid, block, 0, stream>>>(xbf, wcatT, bcat, hg, hb);
    k_biaspart<<<grid, block, 0, stream>>>(hb, xbf, bbiasT, b2b, bacc);
    k_main    <<<grid, block, 0, stream>>>(xbf, hg, w2gbf, bacc, out);
}
